// EdgeDegreeEmbeddingNetwork_eqv2_89979564851622
// MI455X (gfx1250) — matrix operand order repaired and checked by reading, not compiled, not run on hardware
//
#include <hip/hip_runtime.h>
#include <hip/hip_bf16.h>

typedef __attribute__((ext_vector_type(16))) _Float16 v16h;
typedef __attribute__((ext_vector_type(8)))  _Float16 v8h;
typedef __attribute__((ext_vector_type(2)))  __fp16   v2hp;   // cvt_pkrtz result type
typedef __attribute__((ext_vector_type(8)))  float    v8f;
typedef __attribute__((ext_vector_type(4)))  unsigned u32x4;
typedef __attribute__((ext_vector_type(8)))  unsigned u32x8;

#define KNEI   16
#define DIN    320
#define DH     128
#define DOUT   384
#define TILE_M 128
#define STR_X  328   // padded f16 stride for 320-wide rows (mult of 8, breaks bank conflicts)
#define STR_H  136   // padded f16 stride for 128-wide rows
#define LN_EPS 1e-5f
#define INV_AVG (1.0f/23.395238876342773f)
#define SQ3H 0.86602540378f
#define SQ34 0.43301270189f

// LDS layout (bytes)
#define OFF_W  0                 // weightsT f16: up to 384*136 halves = 104448 B
#define OFF_X  104448            // x_edge f16: 128*328 halves = 83968 B (reused for h2)
#define OFF_H  188416            // h1 f16: 128*136 halves = 34816 B
#define OFF_CF 223232            // per-edge coeffs: 128*12 f32 = 6144 B
#define OFF_Z  229376            // zsrc/ztgt ints: 1024 B
#define SMEM_BYTES 230400

// Workspace layout for pre-converted f16 transposed weights (element offsets)
#define WS_W1T 0                         // 128 x STR_X
#define WS_W2T (DH*STR_X)                // 128 x STR_H   (41984)
#define WS_W3T (DH*STR_X + DH*STR_H)     // 384 x STR_H   (59392)

union Frag { v16h v; v8h h[2]; };

__device__ __forceinline__ float hashf(unsigned x) {
    x ^= x >> 17; x *= 0xed5ad4bbu; x ^= x >> 11; x *= 0xac4c1b51u;
    x ^= x >> 15; x *= 0x31848babu; x ^= x >> 14;
    return (float)(x >> 8) * (1.0f/16777216.0f) - 0.5f;
}

// ---- Tensor Data Mover: 2D tile of f16 (data_size=2B) global -> LDS ----
// D# per CDNA5 ISA. group0: count=1, lds_addr[63:32],
// global_addr[120:64], type=2 at [127:126]. group1: data_size=1(2B) at [17:16],
// tensor_dim0[79:48], tensor_dim1[111:80], tile_dim0[127:112], tile_dim1[143:128],
// tensor_dim0_stride[207:160]. Groups 2/3 omitted (<=2D tensor -> VADDR2/3 NULL).
__device__ __forceinline__ void tdm_load_2d_f16(unsigned lds_byte_off, const void* gptr,
                                                unsigned dim0, unsigned rows,
                                                unsigned stride0)
{
    unsigned long long ga = (unsigned long long)gptr;
    u32x4 g0;
    g0[0] = 1u;                                            // count=1, user mode
    g0[1] = lds_byte_off;                                  // lds_addr
    g0[2] = (unsigned)(ga & 0xffffffffu);                  // global_addr[31:0]
    g0[3] = (unsigned)((ga >> 32) & 0x1ffffffu) | (2u << 30);  // addr[56:32] | type=2
    u32x8 g1;
    g1[0] = 0x00010000u;                                   // wg_mask=0, data_size=1 (2B)
    g1[1] = (dim0 & 0xffffu) << 16;                        // tensor_dim0[15:0]
    g1[2] = ((dim0 >> 16) & 0xffffu) | ((rows & 0xffffu) << 16);   // dim0[31:16] | dim1[15:0]
    g1[3] = ((rows >> 16) & 0xffffu) | ((dim0 & 0xffffu) << 16);   // dim1[31:16] | tile_dim0
    g1[4] = rows & 0xffffu;                                // tile_dim1 (tile_dim2=0)
    g1[5] = stride0;                                       // tensor_dim0_stride[31:0]
    g1[6] = 0u;                                            // stride0[47:32]=0, dim1_stride lo
    g1[7] = 0u;
    asm volatile("tensor_load_to_lds %0, %1" :: "s"(g0), "s"(g1) : "memory");
}

__device__ __forceinline__ void tdm_wait() {
#if __has_builtin(__builtin_amdgcn_s_wait_tensorcnt)
    __builtin_amdgcn_s_wait_tensorcnt(0);
#else
    asm volatile("s_wait_tensorcnt 0x0" ::: "memory");
#endif
}

// ---- one-time weight prep: f32 -> f16, transposed (out-ch major), padded strides ----
__global__ void prep_weights(const float* __restrict__ w1, const float* __restrict__ w2,
                             const float* __restrict__ w3, _Float16* __restrict__ ws)
{
    int tid = blockIdx.x * blockDim.x + threadIdx.x;
    int nth = gridDim.x * blockDim.x;
    for (int idx = tid; idx < DIN*DH; idx += nth) {        // w1 (320,128)
        int k = idx >> 7, j = idx & 127;
        ws[WS_W1T + j*STR_X + k] = (_Float16)w1[idx];
    }
    for (int idx = tid; idx < DH*DH; idx += nth) {         // w2 (128,128)
        int k = idx >> 7, j = idx & 127;
        ws[WS_W2T + j*STR_H + k] = (_Float16)w2[idx];
    }
    for (int idx = tid; idx < DH*DOUT; idx += nth) {       // w3 (128,384)
        int k = idx / DOUT, j = idx - k*DOUT;
        ws[WS_W3T + j*STR_H + k] = (_Float16)w3[idx];
    }
}

// LayerNorm + SiLU epilogue on a wave's 16x128 D-slab, write f16 to LDS.
// acc[n][r]: row m = mBase + half*8 + r, channel c = n*16 + lm (WMMA D layout).
__device__ __forceinline__ void ln_silu_store(
    v8f* acc, const float* __restrict__ b, const float* __restrict__ g,
    const float* __restrict__ be, _Float16* dst, int stride,
    int mBase, int lm, int half)
{
    float bb[8], gg[8], bee[8];
#pragma unroll
    for (int n = 0; n < 8; ++n) { int c = n*16 + lm; bb[n] = b[c]; gg[n] = g[c]; bee[n] = be[c]; }
#pragma unroll
    for (int r = 0; r < 8; ++r) {
        float v[8]; float s = 0.f;
#pragma unroll
        for (int n = 0; n < 8; ++n) { v[n] = acc[n][r] + bb[n]; s += v[n]; }
        s += __shfl_xor(s, 1, 32); s += __shfl_xor(s, 2, 32);
        s += __shfl_xor(s, 4, 32); s += __shfl_xor(s, 8, 32);
        float mean = s * (1.f/128.f);
        float q = 0.f;
#pragma unroll
        for (int n = 0; n < 8; ++n) { float d = v[n] - mean; q += d*d; }
        q += __shfl_xor(q, 1, 32); q += __shfl_xor(q, 2, 32);
        q += __shfl_xor(q, 4, 32); q += __shfl_xor(q, 8, 32);
        float inv = rsqrtf(q * (1.f/128.f) + LN_EPS);
        int row = mBase + half*8 + r;
#pragma unroll
        for (int n = 0; n < 8; ++n) {
            float x = (v[n] - mean) * inv * gg[n] + bee[n];
            x = x / (1.f + __expf(-x));                 // SiLU
            dst[row*stride + n*16 + lm] = (_Float16)x;
        }
    }
}

__global__ void __launch_bounds__(256, 1)
edge_degree_embed_fused(const float* __restrict__ edge_scalars,
                        const float* __restrict__ edge_vec,
                        const int*   __restrict__ atomic_numbers,
                        const int*   __restrict__ neighbor_idx,
                        const unsigned char* __restrict__ attn_mask,
                        const float* __restrict__ src_emb,
                        const float* __restrict__ tgt_emb,
                        const _Float16* __restrict__ wsT,
                        const float* __restrict__ b1,
                        const float* __restrict__ g1, const float* __restrict__ be1,
                        const float* __restrict__ b2,
                        const float* __restrict__ g2, const float* __restrict__ be2,
                        const float* __restrict__ b3,
                        float* __restrict__ out)
{
    extern __shared__ __attribute__((aligned(16))) char smem[];
    _Float16* sW   = (_Float16*)(smem + OFF_W);
    _Float16* sX   = (_Float16*)(smem + OFF_X);   // x_edge f16, later h2 f16
    _Float16* sH   = (_Float16*)(smem + OFF_H);   // h1 f16
    float*    cf   = (float*)   (smem + OFF_CF);  // [128][12] wigner coeffs * mask
    int*      zsrc = (int*)     (smem + OFF_Z);
    int*      ztgt = zsrc + TILE_M;

    const int tid   = threadIdx.x;
    const int lane  = tid & 31;
    const int wv    = tid >> 5;
    const int half  = lane >> 4;
    const int lm    = lane & 15;
    const int mBase = wv * 16;
    const int eBase = blockIdx.x * TILE_M;
    const unsigned ldsW = __builtin_amdgcn_groupstaticsize() + OFF_W;

    // ---- kick off TDM for w1^T while we stage activations ----
    if (wv == 0) {
        tdm_load_2d_f16(ldsW, wsT + WS_W1T, STR_X, DH, STR_X);
    }

    // ---- phase 0a: gather atomic numbers for src/tgt embeddings ----
    if (tid < TILE_M) {
        int e = eBase + tid;
        zsrc[tid] = atomic_numbers[neighbor_idx[e]];
        ztgt[tid] = atomic_numbers[e >> 4];        // node = e / KNEI
    }
    __syncthreads();

    // ---- phase 0b: stage x_edge (f32 -> packed f16), per-edge Wigner coeffs ----
    for (int idx = tid; idx < TILE_M * (DIN/4); idx += 256) {
        int row = idx / (DIN/4);
        int ch  = idx - row * (DIN/4);
        const float* sp;
        if (ch < 16)      sp = edge_scalars + (size_t)(eBase + row)*64 + ch*4;
        else if (ch < 48) sp = src_emb + (size_t)zsrc[row]*128 + (ch-16)*4;
        else              sp = tgt_emb + (size_t)ztgt[row]*128 + (ch-48)*4;
        float4 v = *(const float4*)sp;
        _Float16* d = sX + row*STR_X + ch*4;
        *(v2hp*)(d)     = __builtin_amdgcn_cvt_pkrtz(v.x, v.y);
        *(v2hp*)(d + 2) = __builtin_amdgcn_cvt_pkrtz(v.z, v.w);
    }
    if (tid < TILE_M) {
        int e = eBase + tid;
        float vx = edge_vec[e*3+0], vy = edge_vec[e*3+1], vz = edge_vec[e*3+2];
        float nrm = rsqrtf(vx*vx + vy*vy + vz*vz + 1e-30f);
        float nxx = vx*nrm, nxy = vy*nrm, nxz = vz*nrm;
        // pseudo-random v2 (deterministic hash stand-in for jax PRNG key 42)
        unsigned ue = (unsigned)e;
        float ax = hashf(ue*3u + 0x9e3779b9u);
        float ay = hashf(ue*3u + 0x85ebca6bu);
        float az = hashf(ue*3u + 0xc2b2ae35u);
        float r2 = rsqrtf(ax*ax + ay*ay + az*az + 1e-30f);
        float v2x = ax*r2, v2y = ay*r2, v2z = az*r2;
        float bx = -v2y, by = v2x,  bz = v2z;
        float cx =  v2x, cy = -v2z, cz = v2y;
        float dv = fabsf(v2x*nxx + v2y*nxy + v2z*nxz);
        float db = fabsf(bx*nxx + by*nxy + bz*nxz);
        if (dv > db) { v2x = bx; v2y = by; v2z = bz; dv = db; }
        float dc = fabsf(cx*nxx + cy*nxy + cz*nxz);
        if (dv > dc) { v2x = cx; v2y = cy; v2z = cz; }
        // norm_z = normalize(cross(nx, v2)); norm_y = normalize(cross(nx, nz))
        float zx = nxy*v2z - nxz*v2y, zy = nxz*v2x - nxx*v2z, zz = nxx*v2y - nxy*v2x;
        float rz = rsqrtf(zx*zx + zy*zy + zz*zz + 1e-30f); zx*=rz; zy*=rz; zz*=rz;
        float yx = nxy*zz - nxz*zy, yy = nxz*zx - nxx*zz, yz = nxx*zy - nxy*zx;
        float ry = rsqrtf(yx*yx + yy*yy + yz*yz + 1e-30f); yx*=ry; yy*=ry; yz*=ry;
        // R rows: r0=nz, r1=nx, r2=-ny   (R = rot_inv^T)
        float R00 = zx,  R01 = zy,  R02 = zz;
        float R20 = -yx, R21 = -yy, R22 = -yz;
        // x = column 1 of R, normalized + clipped
        float xvx = R01, xvy = nxy, xvz = R21;
        float rx = rsqrtf(xvx*xvx + xvy*xvy + xvz*xvz + 1e-30f);
        float xn0 = fminf(fmaxf(xvx*rx, -1.f), 1.f);
        float xn1 = fminf(fmaxf(xvy*rx, -1.f), 1.f);
        float xn2 = fminf(fmaxf(xvz*rx, -1.f), 1.f);
        float alpha = atan2f(xn0, xn2);
        float sa, ca; __sincosf(alpha, &sa, &ca);
        float cb = xn1;
        float sb = sqrtf(fmaxf(1.f - cb*cb, 0.f));   // beta in [0,pi]
        float Rp00 = ca*R00 - sa*R20;
        float Rp02 = ca*R02 - sa*R22;
        float gamma = atan2f(Rp02, Rp00);
        float sg, cg; __sincosf(gamma, &sg, &cg);
        float c2b = 2.f*cb*cb - 1.f, s2b = 2.f*sb*cb;
        float c2g = cg*cg - sg*sg,   s2g = 2.f*sg*cg;
        // closed-form Wigner rows (alpha cancels in the rows used)
        float d1x = sb*sg, d1y = cb, d1z = -sb*cg;
        float w2c = 0.25f + 0.75f*c2b;
        float w3c = -SQ3H * s2b;
        float w4c =  SQ34 * (1.f - c2b);
        float m = attn_mask[e] ? 0.0f : 1.0f;
        float* c = cf + tid*12;
        c[0] = m;
        c[1] = m*d1x; c[2] = m*d1y; c[3] = m*d1z;
        c[4] = m*(-w4c*s2g); c[5] = m*(-w3c*sg); c[6] = m*w2c;
        c[7] = m*( w3c*cg);  c[8] = m*( w4c*c2g);
        c[9] = 0.f; c[10] = 0.f; c[11] = 0.f;
    }
    if (wv == 0) tdm_wait();           // w1^T resident in LDS
    __syncthreads();

    // ---- GEMM1: [128x320]f16 @ w1[320x128] -> LN -> SiLU -> sH ----
    {
        v8f acc[8] = {};
        for (int kb = 0; kb < DIN; kb += 32) {
            Frag a;
            const _Float16* ap = sX + (mBase + lm)*STR_X + kb + half*8;
            a.h[0] = *(const v8h*)ap;
            a.h[1] = *(const v8h*)(ap + 16);
#pragma unroll
            for (int n = 0; n < 8; ++n) {
                Frag bf;
                const _Float16* bp = sW + (n*16 + lm)*STR_X + kb + half*8;
                bf.h[0] = *(const v8h*)bp;
                bf.h[1] = *(const v8h*)(bp + 16);
                acc[n] = __builtin_amdgcn_wmma_f32_16x16x32_f16(
                    false, a.v, false, bf.v, (short)0, acc[n], false, false);
            }
        }
        ln_silu_store(acc, b1, g1, be1, sH, STR_H, mBase, lm, half);
    }
    __syncthreads();                   // all sW reads done

    // ---- TDM w2^T into sW ----
    if (wv == 0) {
        tdm_load_2d_f16(ldsW, wsT + WS_W2T, STR_H, DH, STR_H);
        tdm_wait();
    }
    __syncthreads();

    // ---- GEMM2: [128x128] @ w2 -> LN -> SiLU -> h2 (reuse sX buffer) ----
    {
        v8f acc[8] = {};
        for (int kb = 0; kb < DH; kb += 32) {
            Frag a;
            const _Float16* ap = sH + (mBase + lm)*STR_H + kb + half*8;
            a.h[0] = *(const v8h*)ap;
            a.h[1] = *(const v8h*)(ap + 16);
#pragma unroll
            for (int n = 0; n < 8; ++n) {
                Frag bf;
                const _Float16* bp = sW + (n*16 + lm)*STR_H + kb + half*8;
                bf.h[0] = *(const v8h*)bp;
                bf.h[1] = *(const v8h*)(bp + 16);
                acc[n] = __builtin_amdgcn_wmma_f32_16x16x32_f16(
                    false, a.v, false, bf.v, (short)0, acc[n], false, false);
            }
        }
        ln_silu_store(acc, b2, g2, be2, sX, STR_H, mBase, lm, half);
    }
    __syncthreads();                   // all sW reads done

    // ---- TDM w3^T (384 x 128) into sW ----
    if (wv == 0) {
        tdm_load_2d_f16(ldsW, wsT + WS_W3T, STR_H, DOUT, STR_H);
        tdm_wait();
    }
    __syncthreads();

    // ---- GEMM3 per 16-col tile + fused Wigner combine + neighbor reduction ----
    // Wave wv owns node = blockIdx*8 + wv (its 16 M-rows are that node's 16 edges).
    const int node = blockIdx.x * 8 + wv;
    float* outp = out + (size_t)node * 1152;     // 9*128 per node
    const float* cfw = cf + (wv*16 + half*8)*12;
#pragma unroll 1
    for (int t = 0; t < 24; ++t) {
        v8f acc = {};
        for (int kb = 0; kb < DH; kb += 32) {
            Frag a;
            const _Float16* ap = sX + (mBase + lm)*STR_H + kb + half*8;
            a.h[0] = *(const v8h*)ap;
            a.h[1] = *(const v8h*)(ap + 16);
            Frag bf;
            const _Float16* bp = sW + (t*16 + lm)*STR_H + kb + half*8;
            bf.h[0] = *(const v8h*)bp;
            bf.h[1] = *(const v8h*)(bp + 16);
            acc = __builtin_amdgcn_wmma_f32_16x16x32_f16(
                false, a.v, false, bf.v, (short)0, acc, false, false);
        }
        float b3c = b3[t*16 + lm];
        float vals[8];
#pragma unroll
        for (int r = 0; r < 8; ++r) vals[r] = acc[r] + b3c;
        int g = t >> 3, cblk = t & 7;
        int istart = (g == 0) ? 0 : ((g == 1) ? 1 : 4);
        int icount = (g == 0) ? 1 : ((g == 1) ? 3 : 5);
        for (int ii = 0; ii < icount; ++ii) {
            int i = istart + ii;
            float p = 0.f;
#pragma unroll
            for (int r = 0; r < 8; ++r) p += cfw[r*12 + i] * vals[r];
            p += __shfl_xor(p, 16, 32);        // combine the two row-halves
            if (half == 0)
                outp[i*128 + cblk*16 + lm] = p * INV_AVG;
        }
    }
}

extern "C" void kernel_launch(void* const* d_in, const int* in_sizes, int n_in,
                              void* d_out, int out_size, void* d_ws, size_t ws_size,
                              hipStream_t stream)
{
    (void)n_in; (void)out_size; (void)ws_size;
    // setup_inputs order: node_input(0, unused), edge_scalars(1), edge_vec(2),
    // atomic_numbers(3), neighbor_idx(4), attn_mask(5), src_emb(6), tgt_emb(7),
    // w1(8), b1(9), g1(10), be1(11), w2(12), b2(13), g2(14), be2(15), w3(16), b3(17)
    const float* edge_scalars = (const float*)d_in[1];
    const float* edge_vec     = (const float*)d_in[2];
    const int*   atomic_num   = (const int*)  d_in[3];
    const int*   neighbor_idx = (const int*)  d_in[4];
    const unsigned char* mask = (const unsigned char*)d_in[5];
    const float* src_emb      = (const float*)d_in[6];
    const float* tgt_emb      = (const float*)d_in[7];
    const float* w1  = (const float*)d_in[8];
    const float* b1  = (const float*)d_in[9];
    const float* g1  = (const float*)d_in[10];
    const float* be1 = (const float*)d_in[11];
    const float* w2  = (const float*)d_in[12];
    const float* b2  = (const float*)d_in[13];
    const float* g2  = (const float*)d_in[14];
    const float* be2 = (const float*)d_in[15];
    const float* w3  = (const float*)d_in[16];
    const float* b3  = (const float*)d_in[17];
    float* out = (float*)d_out;
    _Float16* wsT = (_Float16*)d_ws;   // needs 223232 B

    // One-time (per launch) weight convert/transpose/pad into workspace.
    prep_weights<<<dim3(128), dim3(256), 0, stream>>>(w1, w2, w3, wsT);

    int N = in_sizes[4] / KNEI;       // neighbor_idx is (N, K)
    int blocks = N / 8;               // 8 nodes (128 edges) per block
    edge_degree_embed_fused<<<dim3(blocks), dim3(256), SMEM_BYTES, stream>>>(
        edge_scalars, edge_vec, atomic_num, neighbor_idx, mask,
        src_emb, tgt_emb, wsT, b1, g1, be1, b2, g2, be2, b3, out);
}
